// TransformerSenderGS_8710193676681
// MI455X (gfx1250) — matrix operand order repaired and checked by reading, not compiled, not run on hardware
//
#include <hip/hip_runtime.h>
#include <hip/hip_bf16.h>
#include <math.h>

// ---------------- constants ----------------
#define E_   512
#define V_   8192
#define B_   32
#define T_   16
#define L_   4
#define FF_  2048
#define NH_  8
#define HD_  64

typedef __attribute__((ext_vector_type(16))) __bf16 v16bf;
typedef __attribute__((ext_vector_type(8)))  __bf16 v8bf;
typedef __attribute__((ext_vector_type(8)))  float  v8f;

// ---------------- generic WMMA GEMM ----------------
// C[M,N] = act(alpha * A[M,K] @ W[N,K]^T + bias[N])
// A,W bf16 row-major; Cf fp32 out (optional), Cb bf16 out (optional).
// One wave computes a 16(M) x 64(N) tile: 4 accumulators, A fragment reused 4x.
// N must be a multiple of 64, K a multiple of 32.
__global__ void k_gemm_wmma(const __bf16* __restrict__ A, const __bf16* __restrict__ W,
                            const float* __restrict__ bias,
                            float* __restrict__ Cf, __bf16* __restrict__ Cb,
                            int M, int N, int K, float alpha, int act)
{
    const int wave = (blockIdx.x * blockDim.x + threadIdx.x) >> 5;
    const int lane = threadIdx.x & 31;
    const int nGroups = N >> 6;                 // groups of 4 n-tiles
    const int tg = wave % nGroups;
    const int tm = wave / nGroups;
    if (tm * 16 >= M) return;                   // wave-uniform: EXEC stays all-1s for WMMA

    const int kh  = lane >> 4;                  // 0 or 1 (lane half)
    const int l15 = lane & 15;

    // A fragment (16-bit A 16x32): lane l -> row m=l&15; VGPR0-3: K=kh*8..+7, VGPR4-7: K=16+kh*8..+7
    const __bf16* arow = A + (size_t)(tm * 16 + l15) * K + kh * 8;
    // B fragment (16-bit B 32x16): lane l -> col n=l&15; same K order as A: kh*8..+7, 16+kh*8..+7
    const __bf16* brow[4];
#pragma unroll
    for (int i = 0; i < 4; ++i)
        brow[i] = W + (size_t)(tg * 64 + i * 16 + l15) * K + kh * 8;

    v8f acc[4];
#pragma unroll
    for (int i = 0; i < 4; ++i) acc[i] = (v8f){0.f,0.f,0.f,0.f,0.f,0.f,0.f,0.f};

    for (int k0 = 0; k0 < K; k0 += 32) {
        v8bf alo = *(const v8bf*)(arow + k0);
        v8bf ahi = *(const v8bf*)(arow + k0 + 16);
        v16bf av;
#pragma unroll
        for (int j = 0; j < 8; ++j) { av[j] = alo[j]; av[j + 8] = ahi[j]; }
#pragma unroll
        for (int i = 0; i < 4; ++i) {
            v8bf blo = *(const v8bf*)(brow[i] + k0);
            v8bf bhi = *(const v8bf*)(brow[i] + k0 + 16);
            v16bf bv;
#pragma unroll
            for (int j = 0; j < 8; ++j) { bv[j] = blo[j]; bv[j + 8] = bhi[j]; }
            acc[i] = __builtin_amdgcn_wmma_f32_16x16x32_bf16(
                         false, av, false, bv, (short)0, acc[i], false, false);
        }
    }

    // epilogue: C/D layout: VGPR v, lanes0-15 -> M=v, lanes16-31 -> M=8+v; col = lane&15
    const int rowBase = tm * 16 + kh * 8;
#pragma unroll
    for (int i = 0; i < 4; ++i) {
        const int n = tg * 64 + i * 16 + l15;
        const float bb = bias ? bias[n] : 0.0f;
#pragma unroll
        for (int v = 0; v < 8; ++v) {
            const int row = rowBase + v;
            float val = acc[i][v] * alpha + bb;
            if (act) val = 0.5f * val * (1.0f + erff(val * 0.70710678118654752f)); // exact-erf GELU
            if (row < M) {
                const size_t idx = (size_t)row * N + n;
                if (Cf) Cf[idx] = val;
                if (Cb) Cb[idx] = (__bf16)val;
            }
        }
    }
}

// ---------------- conversion / transpose / broadcast ----------------
__global__ void k_cvt_bf16(const float* __restrict__ in, __bf16* __restrict__ out, int n)
{
    int i = blockIdx.x * blockDim.x + threadIdx.x;
    if (i < n) out[i] = (__bf16)in[i];
}

__global__ void k_transpose_bf16(const float* __restrict__ in, __bf16* __restrict__ out, int R, int C)
{
    int i = blockIdx.x * blockDim.x + threadIdx.x;   // over R*C
    if (i >= R * C) return;
    int r = i / C, c = i - r * C;
    out[(size_t)c * R + r] = (__bf16)in[i];
}

__global__ void k_bcast_sym(const float* __restrict__ sym, float* __restrict__ hf, __bf16* __restrict__ hb)
{
    int i = blockIdx.x * blockDim.x + threadIdx.x;   // B*E
    if (i >= B_ * E_) return;
    float v = sym[i & (E_ - 1)];
    hf[i] = v; hb[i] = (__bf16)v;
}

// ---------------- single-position attention with KV cache ----------------
// qkv: [B, 3E] (q|k|v). caches: [T][B][E]. ob: bf16 [B,E]. p = current position.
__global__ void k_attn_step(const float* __restrict__ qkv,
                            float* __restrict__ kc, float* __restrict__ vc,
                            __bf16* __restrict__ ob, int p)
{
    const int b = blockIdx.x;            // grid = B, block = 256 (8 waves = 8 heads)
    const int h = threadIdx.x >> 5;
    const int lane = threadIdx.x & 31;
    const float* base = qkv + (size_t)b * (3 * E_) + h * HD_;
    const float* qv = base;
    const float* kv = base + E_;
    const float* vv = base + 2 * E_;

    float* kdst = kc + ((size_t)p * B_ + b) * E_ + h * HD_;
    float* vdst = vc + ((size_t)p * B_ + b) * E_ + h * HD_;
    kdst[lane] = kv[lane];  kdst[lane + 32] = kv[lane + 32];
    vdst[lane] = vv[lane];  vdst[lane + 32] = vv[lane + 32];
    __syncthreads();

    float s = -1e30f;
    if (lane <= p) {
        const float* krow = kc + ((size_t)lane * B_ + b) * E_ + h * HD_;
        float acc = 0.f;
#pragma unroll
        for (int d = 0; d < HD_; ++d) acc += qv[d] * krow[d];
        s = acc * 0.125f;                // 1/sqrt(64)
    }
    float mx = s;
#pragma unroll
    for (int off = 16; off > 0; off >>= 1) mx = fmaxf(mx, __shfl_xor(mx, off, 32));
    float e = (lane <= p) ? __expf(s - mx) : 0.f;
    float sum = e;
#pragma unroll
    for (int off = 16; off > 0; off >>= 1) sum += __shfl_xor(sum, off, 32);
    const float prob = e / sum;

    float o0 = 0.f, o1 = 0.f;
    for (int j = 0; j <= p; ++j) {
        const float pj = __shfl(prob, j, 32);
        const float* vrow = vc + ((size_t)j * B_ + b) * E_ + h * HD_;
        o0 += pj * vrow[lane];
        o1 += pj * vrow[lane + 32];
    }
    __bf16* od = ob + (size_t)b * E_ + h * HD_;
    od[lane]      = (__bf16)o0;
    od[lane + 32] = (__bf16)o1;
}

// ---------------- fused residual-add + LayerNorm (E=512, block 256, grid B) ----------------
__global__ void k_add_ln(const float* __restrict__ a, const float* __restrict__ r,
                         const float* __restrict__ g, const float* __restrict__ bt,
                         float* __restrict__ of, __bf16* __restrict__ obf)
{
    __shared__ float red[256];
    const int b = blockIdx.x, t = threadIdx.x;
    const float* ar = a + (size_t)b * E_;
    const float* rr = r + (size_t)b * E_;
    float y0 = ar[t] + rr[t];
    float y1 = ar[t + 256] + rr[t + 256];
    red[t] = y0 + y1; __syncthreads();
    for (int s = 128; s > 0; s >>= 1) { if (t < s) red[t] += red[t + s]; __syncthreads(); }
    const float mean = red[0] * (1.0f / E_);
    __syncthreads();
    const float d0 = y0 - mean, d1 = y1 - mean;
    red[t] = d0 * d0 + d1 * d1; __syncthreads();
    for (int s = 128; s > 0; s >>= 1) { if (t < s) red[t] += red[t + s]; __syncthreads(); }
    const float inv = rsqrtf(red[0] * (1.0f / E_) + 1e-5f);
    const float o0 = d0 * inv * g[t]       + bt[t];
    const float o1 = d1 * inv * g[t + 256] + bt[t + 256];
    float* od = of + (size_t)b * E_;
    od[t] = o0; od[t + 256] = o1;
    if (obf) { __bf16* ob = obf + (size_t)b * E_; ob[t] = (__bf16)o0; ob[t + 256] = (__bf16)o1; }
}

// ---------------- softmax(distrib) + gumbel softmax(sample), V=8192 ----------------
// dist: pre-offset to [step][.][.] of [T,B,V]; seq: d_out base + step*V (row stride (T+1)*V)
__global__ void k_softmax_gumbel(const float* __restrict__ logits, const float* __restrict__ gum,
                                 float* __restrict__ dist, float* __restrict__ seq,
                                 __bf16* __restrict__ sampb)
{
    __shared__ float red[256];
    const int b = blockIdx.x, t = threadIdx.x;
    const float* lr = logits + (size_t)b * V_;
    const float* gr = gum    + (size_t)b * V_;

    float m0 = -1e30f, m1 = -1e30f;
    for (int i = t; i < V_; i += 256) { float l = lr[i]; m0 = fmaxf(m0, l); m1 = fmaxf(m1, l + gr[i]); }
    red[t] = m0; __syncthreads();
    for (int s = 128; s > 0; s >>= 1) { if (t < s) red[t] = fmaxf(red[t], red[t + s]); __syncthreads(); }
    m0 = red[0]; __syncthreads();
    red[t] = m1; __syncthreads();
    for (int s = 128; s > 0; s >>= 1) { if (t < s) red[t] = fmaxf(red[t], red[t + s]); __syncthreads(); }
    m1 = red[0]; __syncthreads();

    float s0 = 0.f, s1 = 0.f;
    for (int i = t; i < V_; i += 256) { float l = lr[i]; s0 += __expf(l - m0); s1 += __expf(l + gr[i] - m1); }
    red[t] = s0; __syncthreads();
    for (int s = 128; s > 0; s >>= 1) { if (t < s) red[t] += red[t + s]; __syncthreads(); }
    s0 = red[0]; __syncthreads();
    red[t] = s1; __syncthreads();
    for (int s = 128; s > 0; s >>= 1) { if (t < s) red[t] += red[t + s]; __syncthreads(); }
    s1 = red[0];

    const float i0 = 1.0f / s0, i1 = 1.0f / s1;
    float* dr = dist + (size_t)b * V_;
    float* sr = seq  + (size_t)b * ((T_ + 1) * V_);
    __bf16* sb = sampb + (size_t)b * V_;
    for (int i = t; i < V_; i += 256) {
        const float l = lr[i];
        dr[i] = __expf(l - m0) * i0;
        const float p = __expf(l + gr[i] - m1) * i1;   // TEMP == 1.0
        sr[i] = p;
        sb[i] = (__bf16)p;
    }
}

// ---------------- EOS row: sequence[:, T, :] = onehot(0) ----------------
__global__ void k_eos(float* __restrict__ seq)
{
    int i = blockIdx.x * blockDim.x + threadIdx.x;   // B*V
    if (i >= B_ * V_) return;
    int b = i >> 13, j = i & (V_ - 1);
    seq[((size_t)b * (T_ + 1) + T_) * V_ + j] = (j == 0) ? 1.0f : 0.0f;
}

// ---------------- host orchestration ----------------
static inline void launch_gemm(hipStream_t s, const __bf16* A, const __bf16* W, const float* bias,
                               float* Cf, __bf16* Cb, int M, int N, int K,
                               float alpha = 1.0f, int act = 0)
{
    const int waves   = (N / 64) * ((M + 15) / 16);
    const int threads = waves * 32;
    const int block   = 256;
    const int grid    = (threads + block - 1) / block;
    k_gemm_wmma<<<grid, block, 0, s>>>(A, W, bias, Cf, Cb, M, N, K, alpha, act);
}

static inline void launch_cvt(hipStream_t s, const float* in, __bf16* out, int n)
{
    k_cvt_bf16<<<(n + 255) / 256, 256, 0, s>>>(in, out, n);
}

extern "C" void kernel_launch(void* const* d_in, const int* in_sizes, int n_in,
                              void* d_out, int out_size, void* d_ws, size_t ws_size,
                              hipStream_t stream)
{
    (void)in_sizes; (void)n_in; (void)out_size; (void)ws_size;
    // inputs (setup_inputs order), all fp32
    const float* enc       = (const float*)d_in[0];   // [1,B,E]
    const float* sym       = (const float*)d_in[1];   // [1,1,E]
    const float* embed_w   = (const float*)d_in[2];   // [V,E]
    const float* out_w     = (const float*)d_in[3];   // [V,E]
    const float* out_b     = (const float*)d_in[4];   // [V]
    const float* gumbel    = (const float*)d_in[5];   // [T,B,V]
    const float* sa_qkv_w  = (const float*)d_in[6];   // [L,3E,E]
    const float* sa_qkv_b  = (const float*)d_in[7];   // [L,3E]
    const float* sa_out_w  = (const float*)d_in[8];   // [L,E,E]
    const float* sa_out_b  = (const float*)d_in[9];   // [L,E]
    const float* ca_qkv_w  = (const float*)d_in[10];  // [L,3E,E]
    const float* ca_qkv_b  = (const float*)d_in[11];  // [L,3E]
    const float* ca_out_w  = (const float*)d_in[12];  // [L,E,E]
    const float* ca_out_b  = (const float*)d_in[13];  // [L,E]
    const float* ln1_g = (const float*)d_in[14]; const float* ln1_b = (const float*)d_in[15];
    const float* ln2_g = (const float*)d_in[16]; const float* ln2_b = (const float*)d_in[17];
    const float* ln3_g = (const float*)d_in[18]; const float* ln3_b = (const float*)d_in[19];
    const float* w1 = (const float*)d_in[20]; const float* b1 = (const float*)d_in[21];
    const float* w2 = (const float*)d_in[22]; const float* b2 = (const float*)d_in[23];

    float* outSeq  = (float*)d_out;                               // [B,T+1,V]
    float* outDist = outSeq + (size_t)B_ * (T_ + 1) * V_;         // [T,B,V]

    // --- workspace bump allocator (256B aligned) ---
    char* wsp = (char*)d_ws;
    auto alloc = [&](size_t bytes) -> void* {
        void* p = (void*)wsp; wsp += (bytes + 255) & ~(size_t)255; return p;
    };
    // bf16 weights
    __bf16* sa_qkv_bf = (__bf16*)alloc((size_t)L_ * 3 * E_ * E_ * 2);
    __bf16* sa_out_bf = (__bf16*)alloc((size_t)L_ * E_ * E_ * 2);
    __bf16* w1_bf     = (__bf16*)alloc((size_t)L_ * FF_ * E_ * 2);
    __bf16* w2_bf     = (__bf16*)alloc((size_t)L_ * E_ * FF_ * 2);
    __bf16* out_w_bf  = (__bf16*)alloc((size_t)V_ * E_ * 2);
    __bf16* embT_bf   = (__bf16*)alloc((size_t)E_ * V_ * 2);      // embed_weight transposed [E,V]
    __bf16* ca_wv_bf  = (__bf16*)alloc((size_t)L_ * E_ * E_ * 2);
    __bf16* ca_ow_bf  = (__bf16*)alloc((size_t)L_ * E_ * E_ * 2);
    __bf16* enc_bf    = (__bf16*)alloc((size_t)B_ * E_ * 2);
    // bf16 activations
    __bf16* hb    = (__bf16*)alloc((size_t)B_ * E_ * 2);
    __bf16* ob    = (__bf16*)alloc((size_t)B_ * E_ * 2);
    __bf16* xb    = (__bf16*)alloc((size_t)B_ * E_ * 2);
    __bf16* x2b   = (__bf16*)alloc((size_t)B_ * E_ * 2);
    __bf16* f1b   = (__bf16*)alloc((size_t)B_ * FF_ * 2);
    __bf16* tmpb  = (__bf16*)alloc((size_t)B_ * E_ * 2);
    __bf16* sampb = (__bf16*)alloc((size_t)B_ * V_ * 2);
    // fp32 state
    float* kcache   = (float*)alloc((size_t)L_ * T_ * B_ * E_ * 4);
    float* vcache   = (float*)alloc((size_t)L_ * T_ * B_ * E_ * 4);
    float* ca_const = (float*)alloc((size_t)L_ * B_ * E_ * 4);
    float* h      = (float*)alloc((size_t)B_ * E_ * 4);
    float* qkv    = (float*)alloc((size_t)B_ * 3 * E_ * 4);
    float* saf    = (float*)alloc((size_t)B_ * E_ * 4);
    float* x      = (float*)alloc((size_t)B_ * E_ * 4);
    float* x2     = (float*)alloc((size_t)B_ * E_ * 4);
    float* f2     = (float*)alloc((size_t)B_ * E_ * 4);
    float* logits = (float*)alloc((size_t)B_ * V_ * 4);
    float* tmpf   = (float*)alloc((size_t)B_ * E_ * 4);

    // --- phase 0: weight conversion (fp32 -> bf16), transpose, CA constants ---
    launch_cvt(stream, sa_qkv_w, sa_qkv_bf, L_ * 3 * E_ * E_);
    launch_cvt(stream, sa_out_w, sa_out_bf, L_ * E_ * E_);
    launch_cvt(stream, w1, w1_bf, L_ * FF_ * E_);
    launch_cvt(stream, w2, w2_bf, L_ * E_ * FF_);
    launch_cvt(stream, out_w, out_w_bf, V_ * E_);
    launch_cvt(stream, ca_out_w, ca_ow_bf, L_ * E_ * E_);
    launch_cvt(stream, enc, enc_bf, B_ * E_);
    for (int l = 0; l < L_; ++l)   // v-projection slice of fused qkv (rows 2E..3E)
        launch_cvt(stream, ca_qkv_w + ((size_t)l * 3 * E_ + 2 * E_) * E_,
                   ca_wv_bf + (size_t)l * E_ * E_, E_ * E_);
    k_transpose_bf16<<<(V_ * E_ + 255) / 256, 256, 0, stream>>>(embed_w, embT_bf, V_, E_);

    // cross-attention over a single encoder position: softmax over 1 key == 1,
    // so CA(x) = out_proj(v_proj(enc)) -- a per-layer constant, independent of x.
    for (int l = 0; l < L_; ++l) {
        launch_gemm(stream, enc_bf, ca_wv_bf + (size_t)l * E_ * E_,
                    ca_qkv_b + (size_t)l * 3 * E_ + 2 * E_, tmpf, tmpb, B_, E_, E_);
        launch_gemm(stream, tmpb, ca_ow_bf + (size_t)l * E_ * E_,
                    ca_out_b + (size_t)l * E_, ca_const + (size_t)l * B_ * E_, nullptr, B_, E_, E_);
    }

    // --- phase 1: autoregressive loop with KV cache (exact vs. full recompute) ---
    k_bcast_sym<<<(B_ * E_ + 255) / 256, 256, 0, stream>>>(sym, h, hb);

    const float embed_scale = sqrtf((float)E_);
    for (int step = 0; step < T_; ++step) {
        for (int l = 0; l < L_; ++l) {
            const size_t wE = (size_t)l * E_ * E_;
            // fused self-attention QKV projection of the new position
            launch_gemm(stream, hb, sa_qkv_bf + (size_t)l * 3 * E_ * E_,
                        sa_qkv_b + (size_t)l * 3 * E_, qkv, nullptr, B_, 3 * E_, E_);
            k_attn_step<<<B_, 256, 0, stream>>>(qkv,
                        kcache + (size_t)l * T_ * B_ * E_,
                        vcache + (size_t)l * T_ * B_ * E_, ob, step);
            launch_gemm(stream, ob, sa_out_bf + wE, sa_out_b + (size_t)l * E_,
                        saf, nullptr, B_, E_, E_);
            k_add_ln<<<B_, 256, 0, stream>>>(h, saf, ln1_g + (size_t)l * E_, ln1_b + (size_t)l * E_, x, xb);
            // cross-attention = precomputed constant
            k_add_ln<<<B_, 256, 0, stream>>>(x, ca_const + (size_t)l * B_ * E_,
                        ln2_g + (size_t)l * E_, ln2_b + (size_t)l * E_, x2, x2b);
            // FFN with fused exact-erf GELU
            launch_gemm(stream, x2b, w1_bf + (size_t)l * FF_ * E_, b1 + (size_t)l * FF_,
                        nullptr, f1b, B_, FF_, E_, 1.0f, 1);
            launch_gemm(stream, f1b, w2_bf + (size_t)l * E_ * FF_, b2 + (size_t)l * E_,
                        f2, nullptr, B_, E_, FF_);
            k_add_ln<<<B_, 256, 0, stream>>>(x2, f2, ln3_g + (size_t)l * E_, ln3_b + (size_t)l * E_, h, hb);
        }
        // vocab logits for the last position only
        launch_gemm(stream, hb, out_w_bf, out_b, logits, nullptr, B_, V_, E_);
        k_softmax_gumbel<<<B_, 256, 0, stream>>>(logits, gumbel + (size_t)step * B_ * V_,
                        outDist + (size_t)step * B_ * V_, outSeq + (size_t)step * V_, sampb);
        // soft embedding of the sample feeds the next position
        launch_gemm(stream, sampb, embT_bf, nullptr, h, hb, B_, E_, V_, embed_scale, 0);
    }

    k_eos<<<(B_ * V_ + 255) / 256, 256, 0, stream>>>(outSeq);
}
